// MultiheadBlockAttention_4715874091496
// MI455X (gfx1250) — matrix operand order repaired and checked by reading, not compiled, not run on hardware
//
#include <hip/hip_runtime.h>

// MI455X / gfx1250, wave32. All matmuls via v_wmma_f32_16x16x32_bf16.
// B=2, S=1024, NB=64, D=256, NHEAD=8, dh=32.

typedef __bf16 bf16;
typedef __attribute__((ext_vector_type(16))) __bf16 v16bf;
typedef __attribute__((ext_vector_type(8)))  __bf16 v8bf;
typedef __attribute__((ext_vector_type(4)))  __bf16 v4bf;
typedef __attribute__((ext_vector_type(8)))  float  v8f;

union V16 { v16bf v; v8bf h[2]; };

#define XSTRIDE 264   // 256 + 8 bf16 pad  -> bank-conflict-free row accesses
#define VSTRIDE 72    // 64 + 8 pad (tokens) for V^T and P rows
#define LDS_Q   0
#define LDS_K   (64 * XSTRIDE)                 // 16896
#define LDS_VT  (2 * 64 * XSTRIDE)             // 33792  (256 feat x 72)
#define LDS_P   (2 * 64 * XSTRIDE + 256 * VSTRIDE)  // 52224 (8 heads x 64 x 72)
#define LDS_ELEMS (LDS_P + 8 * 64 * VSTRIDE)   // 89088 bf16 = 178176 bytes

// ---------------------------------------------------------------------------
// Kernel 1: convert Wq|Wk|Wv (each 256x256 f32, row-major) to bf16 in d_ws.
// ---------------------------------------------------------------------------
__global__ void cvt_weights_kernel(const float* __restrict__ Wq,
                                   const float* __restrict__ Wk,
                                   const float* __restrict__ Wv,
                                   bf16* __restrict__ out) {
  int i = blockIdx.x * blockDim.x + threadIdx.x;      // 0..49151, 4 elems each
  int mat = i / 16384;
  int idx = (i - mat * 16384) * 4;
  const float* src = (mat == 0) ? Wq : (mat == 1) ? Wk : Wv;
  float4 f = *(const float4*)(src + idx);
  v4bf o = { (bf16)f.x, (bf16)f.y, (bf16)f.z, (bf16)f.w };
  *(v4bf*)(out + (size_t)mat * 65536 + idx) = o;
}

// ---------------------------------------------------------------------------
// Kernel 2: one workgroup (256 thr = 8 waves) per (b,s) token position.
// ---------------------------------------------------------------------------
__launch_bounds__(256, 1)
__global__ void block_attn_kernel(const float* __restrict__ q_in,
                                  const float* __restrict__ k_in,
                                  const float* __restrict__ v_in,
                                  const bf16*  __restrict__ wts,   // 3 x 256 x 256 bf16
                                  const float* __restrict__ bq,
                                  const float* __restrict__ bk,
                                  const float* __restrict__ bv,
                                  float* __restrict__ out_attn,    // [B,S,64,256]
                                  float* __restrict__ out_w) {     // [B,8,S,64,64]
  extern __shared__ char smem_raw[];
  bf16* sm  = (bf16*)smem_raw;
  bf16* Qs  = sm + LDS_Q;
  bf16* Ks  = sm + LDS_K;
  bf16* Vts = sm + LDS_VT;   // transposed: [feature=h*32+d][token m], stride 72
  bf16* Ps  = sm + LDS_P;    // per head:   [row n][col m], stride 72
  bf16* Xs  = Ps;            // X staging aliases P region (disjoint in time)

  const int tid  = threadIdx.x;
  const int lane = tid & 31;
  const int wave = tid >> 5;
  const int half = lane >> 4;      // which 16-lane half
  const int lm   = lane & 15;

  const size_t blk = blockIdx.x;                 // (b*1024 + s)
  const float* xin0 = q_in + blk * 16384;
  const float* xin1 = k_in + blk * 16384;
  const float* xin2 = v_in + blk * 16384;

  const int mt  = wave & 3;        // M-tile (token rows) owned by this wave
  const int ntb = (wave >> 2) * 8; // first of 8 N-tiles owned by this wave

  // ======================= Phase A: Q/K/V projections ======================
#pragma unroll
  for (int p = 0; p < 3; ++p) {
    const float* xp = (p == 0) ? xin0 : (p == 1) ? xin1 : xin2;
    const float* bp = (p == 0) ? bq   : (p == 1) ? bk   : bv;

    // stage X[64][256] f32 -> bf16 LDS (row stride 264)
#pragma unroll
    for (int it = 0; it < 16; ++it) {
      int g   = it * 1024 + tid * 4;
      float4 f = *(const float4*)(xp + g);
      int row = g >> 8, col = g & 255;
      v4bf o = { (bf16)f.x, (bf16)f.y, (bf16)f.z, (bf16)f.w };
      *(v4bf*)(Xs + row * XSTRIDE + col) = o;
    }
    __syncthreads();

    v8f acc[8];
#pragma unroll
    for (int j = 0; j < 8; ++j)
#pragma unroll
      for (int e = 0; e < 8; ++e) acc[j][e] = 0.f;

    const bf16* W = wts + (size_t)p * 65536;
    const int arow = (mt * 16 + lm) * XSTRIDE;
#pragma unroll
    for (int kc = 0; kc < 8; ++kc) {
      // A operand: row m over K; K = kc*32 + half*16 + {0..15}
      V16 a;
      int ao = arow + kc * 32 + half * 16;
      a.h[0] = *(const v8bf*)(Xs + ao);
      a.h[1] = *(const v8bf*)(Xs + ao + 8);
#pragma unroll
      for (int j = 0; j < 8; ++j) {
        // B operand: lane = output col n -> row n of W; K = kc*32+half*16+{0..15}
        V16 b;
        int bo = ((ntb + j) * 16 + lm) * 256 + kc * 32 + half * 16;
        b.h[0] = *(const v8bf*)(W + bo);
        b.h[1] = *(const v8bf*)(W + bo + 8);
        acc[j] = __builtin_amdgcn_wmma_f32_16x16x32_bf16(
            false, a.v, false, b.v, (short)0, acc[j], false, false);
      }
    }

    const float scale = (p == 0) ? 0.17677669529663687f : 1.0f; // 1/sqrt(dh) folded into Q
#pragma unroll
    for (int j = 0; j < 8; ++j) {
      int n = (ntb + j) * 16 + lm;          // output feature (C layout: col = lane)
      float bias = bp[n];
      if (p < 2) {
        bf16* dst = (p == 0) ? Qs : Ks;     // row-major [token][feature]
#pragma unroll
        for (int vv = 0; vv < 8; ++vv) {
          int m = mt * 16 + half * 8 + vv;  // C layout: row = vgpr + 8*half
          dst[m * XSTRIDE + n] = (bf16)((acc[j][vv] + bias) * scale);
        }
      } else {
        // V stored transposed: Vt[feature][token], packed b128 store
        v8bf o;
#pragma unroll
        for (int vv = 0; vv < 8; ++vv) o[vv] = (bf16)(acc[j][vv] + bias);
        int m0 = mt * 16 + half * 8;
        *(v8bf*)(Vts + n * VSTRIDE + m0) = o;
      }
    }
    __syncthreads();   // X reused next p; Q/K/Vt visible to all for phase B
  }

  // ======================= Phase B: attention, wave = head =================
  const int h = wave;                      // 8 waves <-> 8 heads
  // ---- scores S = Q_h K_h^T (already scaled), 4x4 tiles, single K-step ----
  v8f sc[4][4];
#pragma unroll
  for (int nt = 0; nt < 4; ++nt)
#pragma unroll
    for (int mtile = 0; mtile < 4; ++mtile)
#pragma unroll
      for (int e = 0; e < 8; ++e) sc[nt][mtile][e] = 0.f;

  V16 bmatS[4];
#pragma unroll
  for (int mtile = 0; mtile < 4; ++mtile) {     // B: lane = token col m -> row of K
    int bo = (mtile * 16 + lm) * XSTRIDE + h * 32 + half * 16;
    bmatS[mtile].h[0] = *(const v8bf*)(Ks + bo);
    bmatS[mtile].h[1] = *(const v8bf*)(Ks + bo + 8);
  }
#pragma unroll
  for (int nt = 0; nt < 4; ++nt) {
    V16 a;                                      // A: lane = token row n, K over d
    int ao = (nt * 16 + lm) * XSTRIDE + h * 32 + half * 16;
    a.h[0] = *(const v8bf*)(Qs + ao);
    a.h[1] = *(const v8bf*)(Qs + ao + 8);
#pragma unroll
    for (int mtile = 0; mtile < 4; ++mtile)
      sc[nt][mtile] = __builtin_amdgcn_wmma_f32_16x16x32_bf16(
          false, a.v, false, bmatS[mtile].v, (short)0, sc[nt][mtile], false, false);
  }

  // ---- softmax over m (64 cols) per row, in registers -----------------------
  // C layout: row n = nt*16 + half*8 + v lives across the 16 lanes of a half,
  // so shfl_xor with masks 1..8 reduces exactly one row.
#pragma unroll
  for (int nt = 0; nt < 4; ++nt) {
#pragma unroll
    for (int vv = 0; vv < 8; ++vv) {
      float mx = fmaxf(fmaxf(sc[nt][0][vv], sc[nt][1][vv]),
                       fmaxf(sc[nt][2][vv], sc[nt][3][vv]));
      mx = fmaxf(mx, __shfl_xor(mx, 1));
      mx = fmaxf(mx, __shfl_xor(mx, 2));
      mx = fmaxf(mx, __shfl_xor(mx, 4));
      mx = fmaxf(mx, __shfl_xor(mx, 8));
      float s = 0.f;
#pragma unroll
      for (int mtile = 0; mtile < 4; ++mtile) {
        float e = __expf(sc[nt][mtile][vv] - mx);
        sc[nt][mtile][vv] = e;
        s += e;
      }
      s += __shfl_xor(s, 1);
      s += __shfl_xor(s, 2);
      s += __shfl_xor(s, 4);
      s += __shfl_xor(s, 8);
      float inv = __builtin_amdgcn_rcpf(s);
#pragma unroll
      for (int mtile = 0; mtile < 4; ++mtile) sc[nt][mtile][vv] *= inv;
    }
  }

  // ---- emit attn_weight (f32, global) + P (bf16, LDS row-major) ------------
  {
    size_t b = blk >> 10, s = blk & 1023;
    float* wbase = out_w + (((b * 8 + h) * 1024 + s) << 12);   // *4096
    bf16*  pbase = Ps + h * (64 * VSTRIDE);
#pragma unroll
    for (int nt = 0; nt < 4; ++nt)
#pragma unroll
      for (int mtile = 0; mtile < 4; ++mtile)
#pragma unroll
        for (int vv = 0; vv < 8; ++vv) {
          int n = nt * 16 + half * 8 + vv;
          int m = mtile * 16 + lm;
          float pv = sc[nt][mtile][vv];
          wbase[n * 64 + m] = pv;              // lanes 0-15: 64B contiguous
          pbase[n * VSTRIDE + m] = (bf16)pv;
        }
  }
  // Same-wave cross-lane LDS handoff: DS pipe is in-order per wave; just stop
  // the compiler/scheduler from hoisting the loads above the stores.
  __builtin_amdgcn_wave_barrier();
  asm volatile("s_wait_dscnt 0" ::: "memory");
  __builtin_amdgcn_wave_barrier();

  // ---- out = P @ V_h : 4 n-tiles x 2 d-tiles, 2 K-steps --------------------
  v8f oacc[4][2];
#pragma unroll
  for (int nt = 0; nt < 4; ++nt)
#pragma unroll
    for (int dt = 0; dt < 2; ++dt)
#pragma unroll
      for (int e = 0; e < 8; ++e) oacc[nt][dt][e] = 0.f;

  const bf16* pbase = Ps + h * (64 * VSTRIDE);
#pragma unroll
  for (int kc = 0; kc < 2; ++kc) {
    V16 bmat[2];
#pragma unroll
    for (int dt = 0; dt < 2; ++dt) {           // B: lane = out col d -> row of V^T
      int bo = (h * 32 + dt * 16 + lm) * VSTRIDE + kc * 32 + half * 16;
      bmat[dt].h[0] = *(const v8bf*)(Vts + bo);
      bmat[dt].h[1] = *(const v8bf*)(Vts + bo + 8);
    }
#pragma unroll
    for (int nt = 0; nt < 4; ++nt) {
      V16 a;                                   // A: lane = row n of P, K over m
      int ao = (nt * 16 + lm) * VSTRIDE + kc * 32 + half * 16;
      a.h[0] = *(const v8bf*)(pbase + ao);
      a.h[1] = *(const v8bf*)(pbase + ao + 8);
#pragma unroll
      for (int dt = 0; dt < 2; ++dt)
        oacc[nt][dt] = __builtin_amdgcn_wmma_f32_16x16x32_bf16(
            false, a.v, false, bmat[dt].v, (short)0, oacc[nt][dt], false, false);
    }
  }

  // ---- store attn_output [64][256] f32 -------------------------------------
  float* obase = out_attn + blk * 16384;
#pragma unroll
  for (int nt = 0; nt < 4; ++nt)
#pragma unroll
    for (int dt = 0; dt < 2; ++dt)
#pragma unroll
      for (int vv = 0; vv < 8; ++vv) {
        int n = nt * 16 + half * 8 + vv;
        int f = h * 32 + dt * 16 + lm;
        obase[n * 256 + f] = oacc[nt][dt][vv]; // lanes 0-15: 64B contiguous
      }
}

// ---------------------------------------------------------------------------
extern "C" void kernel_launch(void* const* d_in, const int* in_sizes, int n_in,
                              void* d_out, int out_size, void* d_ws, size_t ws_size,
                              hipStream_t stream) {
  const float* q  = (const float*)d_in[0];
  const float* k  = (const float*)d_in[1];
  const float* v  = (const float*)d_in[2];
  const float* Wq = (const float*)d_in[3];
  const float* bq = (const float*)d_in[4];
  const float* Wk = (const float*)d_in[5];
  const float* bk = (const float*)d_in[6];
  const float* Wv = (const float*)d_in[7];
  const float* bv = (const float*)d_in[8];

  float* out_attn = (float*)d_out;                       // [2,1024,64,256]
  float* out_w    = out_attn + (size_t)2 * 1024 * 64 * 256; // [2,8,1024,64,64]

  bf16* wts = (bf16*)d_ws;                               // 3 * 65536 bf16 = 384KB

  cvt_weights_kernel<<<192, 256, 0, stream>>>(Wq, Wk, Wv, wts);

  const size_t lds_bytes = (size_t)LDS_ELEMS * sizeof(bf16); // 178176 B
  block_attn_kernel<<<2048, 256, lds_bytes, stream>>>(
      q, k, v, wts, bq, bk, bv, out_attn, out_w);
}
